// ClassAttentionBlock_38199439130875
// MI455X (gfx1250) — matrix operand order repaired and checked by reading, not compiled, not run on hardware
//
#include <hip/hip_runtime.h>
#include <hip/hip_bf16.h>

// ---------------- problem constants ----------------
#define BATCH 64
#define CIN 320
#define COUT 640
#define HW 1024           // 32*32
#define GROUPS 10
#define CGI 32            // in-ch per group
#define CGO 64            // out-ch per group (conv1)
#define NH 4
#define DD 160
#define ALLOCC 16
#define QKVC 1920
#define BN_N 65536.0f     // 64*1024

typedef __attribute__((ext_vector_type(16))) __bf16 v16bf;
typedef __attribute__((ext_vector_type(2)))  __bf16 bf16x2;
typedef __attribute__((ext_vector_type(8)))  float  v8f;

#define LDP 36            // padded LDS leading dim for B tiles (72B rows)
#define LDA 32            // unpadded A tile (async-copied, layout preserving)
#define VLD 18            // padded leading dim for transposed V tile (36B rows)

// ---------------- ws layout (bytes) ----------------
static constexpr size_t SZ_CONV = (size_t)BATCH * COUT * HW * 2;
static constexpr size_t SZ_QKV  = (size_t)BATCH * QKVC * HW * 2;
static constexpr size_t SZ_W1P  = (size_t)COUT * 288 * 2;
static constexpr size_t SZ_WQP  = (size_t)QKVC * 64 * 2;
static constexpr size_t OFF_CONV = 0;
static constexpr size_t OFF_QKV  = OFF_CONV + SZ_CONV;
static constexpr size_t OFF_W1P  = OFF_QKV + SZ_QKV;
static constexpr size_t OFF_WQP  = OFF_W1P + SZ_W1P;
static constexpr size_t OFF_STAT = OFF_WQP + SZ_WQP;
static constexpr size_t OFF_SS   = OFF_STAT + (size_t)COUT * 2 * 4;

static constexpr size_t O_ELEMS    = (size_t)BATCH * COUT * HW;
static constexpr size_t ATTN_ELEMS = (size_t)BATCH * NH * DD * DD;

// ---------------- CDNA5 async copy + WMMA helpers ----------------
__device__ inline unsigned lds_off(const void* p) {
  // generic -> LDS address space -> 32-bit LDS byte offset
  return (unsigned)(size_t)(__attribute__((address_space(3))) const char*)p;
}

__device__ inline void async_g2l_b128(unsigned ldsoff, const void* g) {
  asm volatile("global_load_async_to_lds_b128 %0, %1, off"
               :: "v"(ldsoff), "v"((unsigned long long)(size_t)g)
               : "memory");
}
__device__ inline void async_g2l_b64(unsigned ldsoff, const void* g) {
  asm volatile("global_load_async_to_lds_b64 %0, %1, off"
               :: "v"(ldsoff), "v"((unsigned long long)(size_t)g)
               : "memory");
}
__device__ inline void wait_async0() {
  asm volatile("s_wait_asynccnt 0x0" ::: "memory");
}

__device__ inline v8f vzero8() {
  v8f z;
#pragma unroll
  for (int i = 0; i < 8; ++i) z[i] = 0.0f;
  return z;
}

__device__ inline v8f wmma_bf16(v16bf a, v16bf b, v8f c) {
  return __builtin_amdgcn_wmma_f32_16x16x32_bf16(false, a, false, b, (short)0, c,
                                                 false, false);
}

// A fragment: 16x32 bf16, src row-major [m][k], leading dim ld.
__device__ inline v16bf load_frag_A(const __bf16* a, int ld) {
  const int lane = threadIdx.x & 31;
  const int m  = lane & 15;
  const int k0 = (lane < 16) ? 0 : 8;
  const int k1 = k0 + 16;
  v16bf v;
#pragma unroll
  for (int r = 0; r < 4; ++r) {
    v[2 * r]     = a[m * ld + k0 + 2 * r];
    v[2 * r + 1] = a[m * ld + k0 + 2 * r + 1];
    v[8 + 2 * r]     = a[m * ld + k1 + 2 * r];
    v[8 + 2 * r + 1] = a[m * ld + k1 + 2 * r + 1];
  }
  return v;
}

// B fragment: 32x16 bf16, src stored TRANSPOSED row-major [n][k], leading dim ld.
__device__ inline v16bf load_frag_B(const __bf16* bt, int ld) {
  const int lane = threadIdx.x & 31;
  const int n  = lane & 15;
  const int kb = (lane < 16) ? 0 : 8;
  v16bf v;
#pragma unroll
  for (int r = 0; r < 8; ++r) {
    v[2 * r]     = bt[n * ld + kb + (r >> 2) * 16 + 2 * (r & 3)];
    v[2 * r + 1] = bt[n * ld + kb + (r >> 2) * 16 + 2 * (r & 3) + 1];
  }
  return v;
}

// ---------------- kernel 1: pack weights to bf16 ----------------
__global__ void pack_weights_kernel(const float* __restrict__ w1,
                                    const float* __restrict__ wqkv,
                                    __bf16* __restrict__ w1p,
                                    __bf16* __restrict__ wqp) {
  int i = blockIdx.x * blockDim.x + threadIdx.x;
  if (i < COUT * CGI * 9) {
    int tap = i % 9;
    int t   = i / 9;
    int ci  = t % CGI;
    int cog = t / CGI;
    int g  = cog / CGO;
    int co = cog % CGO;
    w1p[((g * 9 + tap) * CGO + co) * CGI + ci] = (__bf16)w1[i];
  }
  if (i < QKVC * CGO) {
    wqp[i] = (__bf16)wqkv[i];
  }
}

// ---------------- kernel 2: grouped 3x3 conv (implicit GEMM) + bias + BN stats ----------------
__global__ __launch_bounds__(256) void conv_bn_kernel(
    const float* __restrict__ x, const __bf16* __restrict__ w1p,
    const float* __restrict__ b1, __bf16* __restrict__ convo,
    float* __restrict__ stat_sum, float* __restrict__ stat_sq) {
  __shared__ __bf16 ldsA[CGO * LDA];    // [co][ci], async-filled, unpadded
  __shared__ __bf16 ldsB[128 * LDP];    // [pos][ci]
  __shared__ float sSum[CGO];
  __shared__ float sSq[CGO];

  const int tid = threadIdx.x;
  const int gb = blockIdx.y;
  const int b = gb / GROUPS;
  const int g = gb % GROUPS;
  const int hw0 = blockIdx.x * 128;

  if (tid < CGO) { sSum[tid] = 0.0f; sSq[tid] = 0.0f; }

  const int wv = tid >> 5;
  const int lane = tid & 31;
  const int mt = wv & 3;
  const int nc = wv >> 2;

  v8f acc[4];
#pragma unroll
  for (int i = 0; i < 4; ++i) acc[i] = vzero8();

  const float* xg = x + ((size_t)b * CIN + g * CGI) * HW;
  const __bf16* wg = w1p + (size_t)g * 9 * CGO * CGI;
  const unsigned ldsA0 = lds_off(&ldsA[0]);

  // warm the L2/WGP$ path for this block's x tile
  __builtin_prefetch(xg + (size_t)(tid >> 3) * HW + hw0 + (tid & 7) * 16, 0, 3);

  for (int s = 0; s < 9; ++s) {
    __syncthreads();
    // stage A asynchronously: 64x32 bf16 = 4KB, one b128 per thread
    async_g2l_b128(ldsA0 + tid * 16, wg + (size_t)s * CGO * CGI + tid * 8);

    // stage B branchlessly: clamp address, select zero, packed bf16x2 stores
    {
      const int kh = s / 3 - 1;
      const int kw = s % 3 - 1;
      const int pos = tid >> 1;
      const int ci0 = (tid & 1) * 16;
      const int hw = hw0 + pos;
      const int hh = (hw >> 5) + kh;
      const int wwp = (hw & 31) + kw;
      const bool ok = (hh >= 0) && (hh < 32) && (wwp >= 0) && (wwp < 32);
      const int hc = min(31, max(0, hh));
      const int wc = min(31, max(0, wwp));
      const float* xp = xg + hc * 32 + wc;
#pragma unroll
      for (int j = 0; j < 16; j += 2) {
        float v0 = xp[(size_t)(ci0 + j) * HW];
        float v1 = xp[(size_t)(ci0 + j + 1) * HW];
        v0 = ok ? v0 : 0.0f;
        v1 = ok ? v1 : 0.0f;
        bf16x2 p;
        p[0] = (__bf16)v0;
        p[1] = (__bf16)v1;
        *(bf16x2*)&ldsB[pos * LDP + ci0 + j] = p;
      }
    }
    wait_async0();
    __syncthreads();

    v16bf a = load_frag_A(ldsA + mt * 16 * LDA, LDA);
#pragma unroll
    for (int nt = 0; nt < 4; ++nt) {
      v16bf bb = load_frag_B(ldsB + (nc * 64 + nt * 16) * LDP, LDP);
      acc[nt] = wmma_bf16(a, bb, acc[nt]);
    }
  }

  // epilogue: +bias, store bf16, reduce BN stats
  const int mrow = (lane >> 4) << 3;
  const int n = lane & 15;
#pragma unroll
  for (int nt = 0; nt < 4; ++nt) {
#pragma unroll
    for (int r = 0; r < 8; ++r) {
      const int m = mt * 16 + r + mrow;
      const int c = g * CGO + m;
      float v = acc[nt][r] + b1[c];
      const int hw = hw0 + nc * 64 + nt * 16 + n;
      convo[((size_t)b * COUT + c) * HW + hw] = (__bf16)v;
      float rs = v;
      float rq = v * v;
#pragma unroll
      for (int msk = 1; msk < 16; msk <<= 1) {
        rs += __shfl_xor(rs, msk);
        rq += __shfl_xor(rq, msk);
      }
      if (n == 0) {
        atomicAdd(&sSum[m], rs);
        atomicAdd(&sSq[m], rq);
      }
    }
  }
  __syncthreads();
  if (tid < CGO) {
    atomicAdd(&stat_sum[g * CGO + tid], sSum[tid]);
    atomicAdd(&stat_sq[g * CGO + tid], sSq[tid]);
  }
}

// ---------------- kernel 3: BN finalize ----------------
__global__ void bn_finalize_kernel(const float* __restrict__ stat_sum,
                                   const float* __restrict__ stat_sq,
                                   const float* __restrict__ gamma,
                                   const float* __restrict__ beta,
                                   float* __restrict__ bnscale,
                                   float* __restrict__ bnshift) {
  int c = blockIdx.x * blockDim.x + threadIdx.x;
  if (c < COUT) {
    float mean = stat_sum[c] * (1.0f / BN_N);
    float var = stat_sq[c] * (1.0f / BN_N) - mean * mean;
    float sc = gamma[c] * rsqrtf(var + 1e-5f);
    bnscale[c] = sc;
    bnshift[c] = beta[c] - mean * sc;
  }
}

// ---------------- kernel 4: grouped 1x1 qkv conv, BN+ReLU fused ----------------
__global__ __launch_bounds__(256) void qkv_kernel(
    const __bf16* __restrict__ convo, const __bf16* __restrict__ wqp,
    const float* __restrict__ bnscale, const float* __restrict__ bnshift,
    __bf16* __restrict__ qkvo) {
  __shared__ __bf16 ldsA[CGO * LDA];
  __shared__ __bf16 ldsB[128 * LDP];

  const int tid = threadIdx.x;
  const int gb = blockIdx.z;
  const int b = gb / GROUPS;
  const int g = gb % GROUPS;
  const int m64 = blockIdx.y;
  const int hw0 = blockIdx.x * 128;

  const int wv = tid >> 5;
  const int lane = tid & 31;
  const int mt = wv & 3;
  const int nc = wv >> 2;

  v8f acc[4];
#pragma unroll
  for (int i = 0; i < 4; ++i) acc[i] = vzero8();

  const unsigned ldsA0 = lds_off(&ldsA[0]);
  const __bf16* wbase = wqp + (size_t)(g * 192 + m64 * 64) * 64;

  for (int st = 0; st < 2; ++st) {
    __syncthreads();
    // stage A async: thread's 8 bf16 sit inside one 32-wide K block (16B aligned)
    {
      int idx = tid * 8;
      int co = idx >> 5;
      int ci0 = idx & 31;
      async_g2l_b128(ldsA0 + tid * 16, wbase + (size_t)co * 64 + st * 32 + ci0);
    }
    // stage B: one channel x 16 contiguous positions per thread, BN+ReLU fused
    {
      const int ci = tid >> 3;
      const int pos0 = (tid & 7) * 16;
      const int c = g * CGO + st * 32 + ci;
      const float scl = bnscale[c];
      const float sh  = bnshift[c];
      const __bf16* srcrow = convo + ((size_t)b * COUT + c) * HW + hw0 + pos0;
#pragma unroll
      for (int t = 0; t < 16; ++t) {
        float v = (float)srcrow[t] * scl + sh;
        v = fmaxf(v, 0.0f);
        ldsB[(pos0 + t) * LDP + ci] = (__bf16)v;
      }
    }
    wait_async0();
    __syncthreads();

    v16bf a = load_frag_A(ldsA + mt * 16 * LDA, LDA);
#pragma unroll
    for (int nt = 0; nt < 4; ++nt) {
      v16bf bb = load_frag_B(ldsB + (nc * 64 + nt * 16) * LDP, LDP);
      acc[nt] = wmma_bf16(a, bb, acc[nt]);
    }
  }

  const int mrow = (lane >> 4) << 3;
  const int n = lane & 15;
#pragma unroll
  for (int nt = 0; nt < 4; ++nt) {
#pragma unroll
    for (int r = 0; r < 8; ++r) {
      const int m = mt * 16 + r + mrow;
      const int cout = g * 192 + m64 * 64 + m;
      const int hw = hw0 + nc * 64 + nt * 16 + n;
      qkvo[((size_t)b * QKVC + cout) * HW + hw] = (__bf16)acc[nt][r];
    }
  }
}

// ---------------- kernel 5: masked attention ----------------
__global__ __launch_bounds__(320) void attn_kernel(
    const __bf16* __restrict__ qkv, const int* __restrict__ ycls,
    float* __restrict__ o_out, float* __restrict__ attn_out) {
  __shared__ __bf16 ldsKT[16 * LDP];        // [j][s-chunk 32], async-filled
  __shared__ __bf16 ldsVP[HW * VLD];        // transposed V: [s][j], padded
  __shared__ float ldsAttn[10 * 16 * 17];

  const int bh = blockIdx.x;
  const int b = bh >> 2;
  const int h = bh & 3;
  const int tid = threadIdx.x;
  const int wv = tid >> 5;
  const int lane = tid & 31;
  const int yc = ycls[b];

  const size_t qkvb = (size_t)b * QKVC * HW;
  const int kbase = COUT + h * DD + yc * ALLOCC;
  const int vbase = 2 * COUT + h * DD + yc * ALLOCC;
  const unsigned ldsKT0 = lds_off(&ldsKT[0]);

  // stage V transposed: ldsVP[s][j] = v[j][s]
  for (int it = 0; it < 52; ++it) {
    int idx = it * 320 + tid;
    if (idx < 16 * HW) {
      int j = idx >> 10;
      int s = idx & 1023;
      ldsVP[s * VLD + j] = qkv[qkvb + (size_t)(vbase + j) * HW + s];
    }
  }

  // ---- GEMM1: dot[i][j] = sum_s q[i][s] * k[j][s] ----
  const int i0 = wv * 16;
  const int m = lane & 15;
  const int k0 = (lane < 16) ? 0 : 8;
  const int k1 = k0 + 16;
  const unsigned* qrow =
      (const unsigned*)(qkv + qkvb + (size_t)(h * DD + i0 + m) * HW);

  v8f acc = vzero8();
  for (int sc = 0; sc < 32; ++sc) {
    __syncthreads();
    // async-stage the 16x32 K chunk: 128 b64 transfers (8B-aligned rows of 72B)
    if (tid < 128) {
      int j = tid >> 3;
      int q8 = tid & 7;
      async_g2l_b64(ldsKT0 + j * (LDP * 2) + q8 * 8,
                    qkv + qkvb + (size_t)(kbase + j) * HW + sc * 32 + q8 * 4);
    }
    wait_async0();
    __syncthreads();

    union { v16bf v; unsigned u[8]; } A;
    const unsigned* qp = qrow + sc * 16;
#pragma unroll
    for (int r = 0; r < 4; ++r) {
      A.u[r]     = qp[(k0 >> 1) + r];
      A.u[4 + r] = qp[(k1 >> 1) + r];
    }
    v16bf bb = load_frag_B(ldsKT, LDP);
    acc = wmma_bf16(A.v, bb, acc);
  }

  // ---- scale + softmax over 16 active columns ----
  const float SCALE = 0.03952847075210474f;  // 640^-0.5
  const int mrow = (lane >> 4) << 3;
#pragma unroll
  for (int r = 0; r < 8; ++r) {
    float v = acc[r] * SCALE;
    float mx = v;
#pragma unroll
    for (int msk = 1; msk < 16; msk <<= 1) mx = fmaxf(mx, __shfl_xor(mx, msk));
    float ev = __expf(v - mx);
    float sm = ev;
#pragma unroll
    for (int msk = 1; msk < 16; msk <<= 1) sm += __shfl_xor(sm, msk);
    float a = ev / sm;
    const int mi = r + mrow;
    const int irow = i0 + mi;
    attn_out[((size_t)bh * DD + irow) * DD + yc * ALLOCC + (lane & 15)] = a;
    ldsAttn[(wv * 16 + mi) * 17 + (lane & 15)] = a;
  }
  __syncthreads();

  // ---- GEMM2: o[i][s] = sum_j attn[i][j] * v[j][s]  (K=16 zero-padded to 32) ----
  v16bf a2;
  {
    const int kb2 = (lane < 16) ? 0 : 8;
#pragma unroll
    for (int e = 0; e < 8; ++e) {
      a2[e] = (__bf16)ldsAttn[(wv * 16 + m) * 17 + kb2 + e];
      a2[8 + e] = (__bf16)0.0f;
    }
  }
  const int n = lane & 15;
  for (int nt = 0; nt < 64; ++nt) {
    union { v16bf v; unsigned u[8]; } B2;
    const int s = nt * 16 + n;
    const unsigned* vp = (const unsigned*)&ldsVP[s * VLD];
#pragma unroll
    for (int r = 0; r < 8; ++r) {
      unsigned u = (r < 4) ? vp[r + ((lane < 16) ? 0 : 4)] : 0u;  // bf16 pair j=8h+2r, 8h+2r+1
      B2.u[r] = u;   // elements 8..15 carry K=16..31 (zero pad)
    }
    v8f oc = wmma_bf16(a2, B2.v, vzero8());
#pragma unroll
    for (int r = 0; r < 8; ++r) {
      const int mi = r + mrow;
      o_out[((size_t)b * COUT + h * DD + i0 + mi) * HW + nt * 16 + n] = oc[r];
    }
  }
}

// ---------------- launch ----------------
extern "C" void kernel_launch(void* const* d_in, const int* in_sizes, int n_in,
                              void* d_out, int out_size, void* d_ws, size_t ws_size,
                              hipStream_t stream) {
  const float* x     = (const float*)d_in[0];
  const int*   y     = (const int*)d_in[1];
  const float* w1    = (const float*)d_in[2];
  const float* b1    = (const float*)d_in[3];
  const float* gamma = (const float*)d_in[4];
  const float* beta  = (const float*)d_in[5];
  const float* wqkv  = (const float*)d_in[6];

  float* out = (float*)d_out;
  char* ws = (char*)d_ws;

  __bf16* convo = (__bf16*)(ws + OFF_CONV);
  __bf16* qkvo  = (__bf16*)(ws + OFF_QKV);
  __bf16* w1p   = (__bf16*)(ws + OFF_W1P);
  __bf16* wqp   = (__bf16*)(ws + OFF_WQP);
  float* stat_sum = (float*)(ws + OFF_STAT);
  float* stat_sq  = stat_sum + COUT;
  float* bnscale  = (float*)(ws + OFF_SS);
  float* bnshift  = bnscale + COUT;

  float* o_out    = out;
  float* attn_out = out + O_ELEMS;

  (void)hipMemsetAsync(stat_sum, 0, (size_t)COUT * 2 * sizeof(float), stream);
  (void)hipMemsetAsync(attn_out, 0, ATTN_ELEMS * sizeof(float), stream);

  pack_weights_kernel<<<720, 256, 0, stream>>>(w1, wqkv, w1p, wqp);

  conv_bn_kernel<<<dim3(8, BATCH * GROUPS), 256, 0, stream>>>(
      x, w1p, b1, convo, stat_sum, stat_sq);

  bn_finalize_kernel<<<3, 256, 0, stream>>>(stat_sum, stat_sq, gamma, beta,
                                            bnscale, bnshift);

  qkv_kernel<<<dim3(8, 3, BATCH * GROUPS), 256, 0, stream>>>(
      convo, wqp, bnscale, bnshift, qkvo);

  attn_kernel<<<BATCH * NH, 320, 0, stream>>>(qkvo, y, o_out, attn_out);
}
